// TTBlock_69097433858499
// MI455X (gfx1250) — matrix operand order repaired and checked by reading, not compiled, not run on hardware
//
#include <hip/hip_runtime.h>
#include <hip/hip_bf16.h>
#include <math.h>

// ---------------------------------------------------------------------------
// CDNA5 (gfx1250) TT-MLP block:
//   out = (GELU(x @ W0 + b0)) @ W1 + b1
// W0/W1 composed once from TT cores (tiny), stored transposed (N-major x K)
// in bf16; x converted to bf16 once; both GEMMs run pure-bf16 fragments on
// v_wmma_f32_16x16x32_bf16 with fp32 accumulation.
// ---------------------------------------------------------------------------

typedef __attribute__((ext_vector_type(16))) __bf16 bf16x16;
typedef __attribute__((ext_vector_type(8)))  float  f32x8;

union FragAB {
    bf16x16  v;
    uint4    q[2];
};

__device__ __forceinline__ unsigned short f2bf(float x) {
    unsigned u = __float_as_uint(x);
    u = (u + 0x7FFFu + ((u >> 16) & 1u)) >> 16;   // round-to-nearest-even
    return (unsigned short)u;
}

__device__ __forceinline__ unsigned pack2bf(float a, float b) {
    return (unsigned)f2bf(a) | ((unsigned)f2bf(b) << 16);
}

// A fragment (16x32 bf16): lane<16 -> row M=lane,   K = [kb..kb+8) U [kb+16..kb+24)
//                          lane>=16 -> row M=lane-16, base shifted by +8.
__device__ __forceinline__ void load_frag_a(FragAB& f, const unsigned short* A,
                                            int lda, int m, int kb, int lane) {
    const int row = m + (lane & 15);
    const int kk  = kb + ((lane >> 4) << 3);
    const unsigned short* p = A + (size_t)row * lda + kk;
    f.q[0] = *reinterpret_cast<const uint4*>(p);        // K: kk..kk+7
    f.q[1] = *reinterpret_cast<const uint4*>(p + 16);   // K: kk+16..kk+23
}

// B fragment (32x16 bf16), weights stored transposed (N-major x K):
// lane<16 -> col N=lane, K=[kb..kb+8) U [kb+16..kb+24); lane>=16 -> col N=lane-16, base shifted by +8
__device__ __forceinline__ void load_frag_b(FragAB& f, const unsigned short* BT,
                                            int ldk, int n, int kb, int lane) {
    const int col = n + (lane & 15);
    const int kk  = kb + ((lane >> 4) << 3);
    const unsigned short* p = BT + (size_t)col * ldk + kk;
    f.q[0] = *reinterpret_cast<const uint4*>(p);
    f.q[1] = *reinterpret_cast<const uint4*>(p + 16);
}

__device__ __forceinline__ float gelu_exact(float x) {
    return 0.5f * x * (1.0f + erff(x * 0.70710678118654752440f));
}

// ---------------------------------------------------------------------------
// GEMM: C(MxN) = epilogue(A(MxK,bf16) * BT(NxK,bf16)^T + bias)
// Block = 256 threads = 8 waves, wave grid 2(M) x 4(N), wave tile 64x64,
// block tile 128x256.  MODE 0: GELU -> bf16 store.  MODE 1: fp32 store.
// ---------------------------------------------------------------------------
template <int MODE>
__global__ __launch_bounds__(256) void tt_gemm(
    const unsigned short* __restrict__ A, const unsigned short* __restrict__ BT,
    const float* __restrict__ bias, void* __restrict__ Cout, int K, int N)
{
    const int lane  = threadIdx.x & 31;
    const int wave  = threadIdx.x >> 5;
    const int waveM = wave >> 2;   // 0..1
    const int waveN = wave & 3;    // 0..3
    const int M0 = blockIdx.y * 128 + waveM * 64;
    const int N0 = blockIdx.x * 256 + waveN * 64;

    f32x8 acc[4][4];
    const f32x8 zero = {0.f, 0.f, 0.f, 0.f, 0.f, 0.f, 0.f, 0.f};
    #pragma unroll
    for (int i = 0; i < 4; ++i)
        #pragma unroll
        for (int j = 0; j < 4; ++j) acc[i][j] = zero;

    for (int kb = 0; kb < K; kb += 32) {
        FragAB a[4], b[4];
        #pragma unroll
        for (int i = 0; i < 4; ++i) load_frag_a(a[i], A, K, M0 + i * 16, kb, lane);
        #pragma unroll
        for (int j = 0; j < 4; ++j) load_frag_b(b[j], BT, K, N0 + j * 16, kb, lane);

        if (kb + 32 < K) {   // wave-uniform: EXEC stays all-ones
            #pragma unroll
            for (int i = 0; i < 4; ++i)
                __builtin_prefetch(A + (size_t)(M0 + i * 16 + (lane & 15)) * K + kb + 32, 0, 3);
            #pragma unroll
            for (int j = 0; j < 4; ++j)
                __builtin_prefetch(BT + (size_t)(N0 + j * 16 + (lane & 15)) * K + kb + 32, 0, 3);
        }

        #pragma unroll
        for (int i = 0; i < 4; ++i)
            #pragma unroll
            for (int j = 0; j < 4; ++j)
                acc[i][j] = __builtin_amdgcn_wmma_f32_16x16x32_bf16(
                    /*neg_a=*/false, a[i].v, /*neg_b=*/false, b[j].v,
                    /*c_mod=*/(short)0, acc[i][j],
                    /*reuse_a=*/false, /*reuse_b=*/false);
    }

    // C layout: VGPR v -> M = v + 8*(lane>=16), N = lane&15
    const int rbase = (lane >> 4) << 3;
    const int cn    = lane & 15;
    #pragma unroll
    for (int j = 0; j < 4; ++j) {
        const int ncol = N0 + j * 16 + cn;
        const float bv = bias[ncol];
        #pragma unroll
        for (int i = 0; i < 4; ++i) {
            const int mrow = M0 + i * 16 + rbase;
            #pragma unroll
            for (int v = 0; v < 8; ++v) {
                float val = acc[i][j][v] + bv;
                if (MODE == 0) {
                    val = gelu_exact(val);
                    ((unsigned short*)Cout)[(size_t)(mrow + v) * N + ncol] = f2bf(val);
                } else {
                    ((float*)Cout)[(size_t)(mrow + v) * N + ncol] = val;
                }
            }
        }
    }
}

// ---------------------------------------------------------------------------
// x (fp32) -> bf16, 8 elements per thread, vectorized.
// ---------------------------------------------------------------------------
__global__ __launch_bounds__(256) void cvt_bf16(const float* __restrict__ X,
                                                unsigned short* __restrict__ X16)
{
    const size_t t = (size_t)blockIdx.x * 256 + threadIdx.x;
    const float4 a = *reinterpret_cast<const float4*>(X + t * 8);
    const float4 b = *reinterpret_cast<const float4*>(X + t * 8 + 4);
    uint4 o;
    o.x = pack2bf(a.x, a.y);
    o.y = pack2bf(a.z, a.w);
    o.z = pack2bf(b.x, b.y);
    o.w = pack2bf(b.z, b.w);
    *reinterpret_cast<uint4*>(X16 + t * 8) = o;
}

// ---------------------------------------------------------------------------
// Compose W0^T (4096 x 1024, bf16): W0[(i,j,k),(o,p,q)] = sum_{r,s} c0a*c0b*c0c
// in_shape (8,16,8), out_shape (16,16,16), ranks (1,8,8,1)
// ---------------------------------------------------------------------------
__global__ __launch_bounds__(256) void tt_compose_w0(
    const float* __restrict__ ca, const float* __restrict__ cb,
    const float* __restrict__ cc, unsigned short* __restrict__ WT)
{
    const int t  = blockIdx.x * 256 + threadIdx.x;   // 0 .. 4096*1024-1
    const int n  = t >> 10;                          // opq
    const int kk = t & 1023;                         // ijk
    const int o = n >> 8, p = (n >> 4) & 15, q = n & 15;
    const int i = kk >> 7, j = (kk >> 3) & 15, k = kk & 7;
    float acc = 0.f;
    #pragma unroll
    for (int s = 0; s < 8; ++s) {
        float u = 0.f;
        #pragma unroll
        for (int r = 0; r < 8; ++r)
            u += ca[(i * 16 + o) * 8 + r] * cb[((r * 16 + j) * 16 + p) * 8 + s];
        acc += u * cc[(s * 8 + k) * 16 + q];
    }
    WT[(size_t)n * 1024 + kk] = f2bf(acc);
}

// ---------------------------------------------------------------------------
// Compose W1^T (1024 x 4096, bf16): in_shape (16,16,16), out_shape (8,16,8)
// ---------------------------------------------------------------------------
__global__ __launch_bounds__(256) void tt_compose_w1(
    const float* __restrict__ ca, const float* __restrict__ cb,
    const float* __restrict__ cc, unsigned short* __restrict__ WT)
{
    const int t  = blockIdx.x * 256 + threadIdx.x;   // 0 .. 1024*4096-1
    const int n  = t >> 12;                          // opq (0..1023)
    const int kk = t & 4095;                         // ijk (0..4095)
    const int o = n >> 7, p = (n >> 3) & 15, q = n & 7;
    const int i = kk >> 8, j = (kk >> 4) & 15, k = kk & 15;
    float acc = 0.f;
    #pragma unroll
    for (int s = 0; s < 8; ++s) {
        float u = 0.f;
        #pragma unroll
        for (int r = 0; r < 8; ++r)
            u += ca[(i * 8 + o) * 8 + r] * cb[((r * 16 + j) * 16 + p) * 8 + s];
        acc += u * cc[(s * 16 + k) * 8 + q];
    }
    WT[(size_t)n * 4096 + kk] = f2bf(acc);
}

// ---------------------------------------------------------------------------
extern "C" void kernel_launch(void* const* d_in, const int* in_sizes, int n_in,
                              void* d_out, int out_size, void* d_ws, size_t ws_size,
                              hipStream_t stream) {
    (void)in_sizes; (void)n_in; (void)out_size; (void)ws_size;
    const float* x   = (const float*)d_in[0];
    const float* c0a = (const float*)d_in[1];
    const float* c0b = (const float*)d_in[2];
    const float* c0c = (const float*)d_in[3];
    const float* b0  = (const float*)d_in[4];
    const float* c1a = (const float*)d_in[5];
    const float* c1b = (const float*)d_in[6];
    const float* c1c = (const float*)d_in[7];
    const float* b1  = (const float*)d_in[8];

    // Workspace (bf16): X16 16MB | W0T 8MB | W1T 8MB | H 64MB = 96MB total
    unsigned short* X16 = (unsigned short*)d_ws;
    unsigned short* WT0 = X16 + (size_t)8192 * 1024;
    unsigned short* WT1 = WT0 + (size_t)4096 * 1024;
    unsigned short* H   = WT1 + (size_t)1024 * 4096;

    cvt_bf16<<<(8192 * 1024) / (256 * 8), 256, 0, stream>>>(x, X16);
    tt_compose_w0<<<(4096 * 1024) / 256, 256, 0, stream>>>(c0a, c0b, c0c, WT0);
    tt_compose_w1<<<(1024 * 4096) / 256, 256, 0, stream>>>(c1a, c1b, c1c, WT1);

    // GEMM1: H = GELU(x(8192x1024) * W0 + b0) -> bf16 (8192x4096)
    tt_gemm<0><<<dim3(4096 / 256, 8192 / 128), 256, 0, stream>>>(
        X16, WT0, b0, (void*)H, /*K=*/1024, /*N=*/4096);

    // GEMM2: out = H(8192x4096) * W1 + b1 -> fp32 (8192x1024)
    tt_gemm<1><<<dim3(1024 / 256, 8192 / 128), 256, 0, stream>>>(
        H, WT1, b1, d_out, /*K=*/4096, /*N=*/1024);
}
